// SelfAttention_10118942949682
// MI455X (gfx1250) — matrix operand order repaired and checked by reading, not compiled, not run on hardware
//
#include <hip/hip_runtime.h>
#include <hip/hip_bf16.h>

// ---------- problem constants ----------
#define BATCH   8
#define CH      512
#define SPA     1024            // H*W = 32*32
#define GROUPS  32
#define CPG     (CH / GROUPS)   // 16 channels per group
#define HEADS   8
#define HD      (CH / HEADS)    // 64
#define EPSV    1e-5f

typedef __attribute__((ext_vector_type(16))) __bf16 v16bf;
typedef __attribute__((ext_vector_type(8)))  float  v8f;

// ---------- WMMA wrapper ----------
__device__ inline v8f wmma_bf16(v16bf a, v16bf b, v8f c) {
    // D = A(16x32 bf16) * B(32x16 bf16) + C(16x16 f32)
    return __builtin_amdgcn_wmma_f32_16x16x32_bf16(
        /*neg_a=*/false, a, /*neg_b=*/false, b,
        /*c_mod=*/(short)0, c, /*reuse_a=*/false, /*reuse_b=*/false);
}

// ---------- CDNA5 async global->LDS copy (ASYNCcnt path) ----------
// LDS destination address is taken per lane from a VGPR; generic pointer low 32 bits = LDS byte offset.
__device__ inline void async_copy_b128(void* lds_ptr, const void* gptr) {
    uint32_t l = (uint32_t)(uintptr_t)lds_ptr;
    uint64_t g = (uint64_t)(uintptr_t)gptr;
    asm volatile("global_load_async_to_lds_b128 %0, %1, off"
                 :: "v"(l), "v"(g) : "memory");
}
__device__ inline void wait_async0() { asm volatile("s_wait_asynccnt 0x0" ::: "memory"); }
__device__ inline void wait_async1() { asm volatile("s_wait_asynccnt 0x1" ::: "memory"); }
__device__ inline void wait_async2() { asm volatile("s_wait_asynccnt 0x2" ::: "memory"); }

// ---------- fragment loaders (layouts per CDNA5 ISA 7.12.2) ----------
// A 16x32 (MxK), row-major A[m*lda + k]:
//   lane<16 : row=lane, K-base 0 ; lane>=16 : row=lane-16, K-base 8
//   e[0..7] = K+0..7, e[8..15] = K+16..23
__device__ inline v16bf load_A(const __bf16* A, int lda, int m0, int k0, int lane) {
    const __bf16* p = A + (size_t)(m0 + (lane & 15)) * lda + k0 + ((lane & 16) ? 8 : 0);
    v16bf r;
#pragma unroll
    for (int i = 0; i < 8; ++i) r[i] = p[i];
#pragma unroll
    for (int i = 0; i < 8; ++i) r[8 + i] = p[16 + i];
    return r;
}

// A 16x32 from a transposed source: A[m,k] = Q[k*ldq + m]  (Q is [K x M] row-major)
__device__ inline v16bf load_At(const __bf16* Q, int ldq, int m0, int k0, int lane) {
    int m  = m0 + (lane & 15);
    int kb = k0 + ((lane & 16) ? 8 : 0);
    v16bf r;
#pragma unroll
    for (int i = 0; i < 8; ++i) r[i] = Q[(size_t)(kb + i) * ldq + m];
#pragma unroll
    for (int i = 0; i < 8; ++i) r[8 + i] = Q[(size_t)(kb + 16 + i) * ldq + m];
    return r;
}

// ---------- kernel 1: f32 -> bf16 convert ----------
__global__ void cvt_bf16_kernel(const float* __restrict__ src, __bf16* __restrict__ dst, int n) {
    int i = blockIdx.x * blockDim.x + threadIdx.x;
    if (i < n) dst[i] = (__bf16)src[i];
}

// ---------- kernel 2: GroupNorm stats (one block per (b,group)) ----------
__global__ __launch_bounds__(256) void gn_stats_kernel(const float* __restrict__ x,
                                                       float* __restrict__ mean,
                                                       float* __restrict__ rstd) {
    const int grp = blockIdx.x;                 // b*GROUPS + g : contiguous slab of CPG*SPA floats
    const float* p = x + (size_t)grp * (CPG * SPA);
    float s = 0.f, ss = 0.f;
    for (int i = threadIdx.x; i < CPG * SPA; i += blockDim.x) {
        float v = p[i]; s += v; ss += v * v;
    }
    __shared__ float shs[256], shq[256];
    shs[threadIdx.x] = s; shq[threadIdx.x] = ss;
    __syncthreads();
    for (int st = 128; st > 0; st >>= 1) {
        if (threadIdx.x < st) { shs[threadIdx.x] += shs[threadIdx.x + st];
                                shq[threadIdx.x] += shq[threadIdx.x + st]; }
        __syncthreads();
    }
    if (threadIdx.x == 0) {
        const float inv_n = 1.0f / (float)(CPG * SPA);
        float m  = shs[0] * inv_n;
        float var = shq[0] * inv_n - m * m;
        mean[grp] = m;
        rstd[grp] = rsqrtf(var + EPSV);
    }
}

// ---------- kernel 3: GN apply + pack bf16 ----------
__global__ __launch_bounds__(256) void gn_apply_kernel(const float* __restrict__ x,
                                                       const float* __restrict__ mean,
                                                       const float* __restrict__ rstd,
                                                       const float* __restrict__ w,
                                                       const float* __restrict__ bias,
                                                       __bf16* __restrict__ xn) {
    int i = blockIdx.x * blockDim.x + threadIdx.x;   // exact launch over B*C*S
    int c   = (i >> 10) & (CH - 1);
    int grp = i >> 14;                               // (b*CH + c)/CPG
    float v = (x[i] - mean[grp]) * rstd[grp] * w[c] + bias[c];
    xn[i] = (__bf16)v;
}

// ---------- kernel 4: QKV GEMM with double-buffered async B tiles ----------
// C[m, n] = sum_k W[m,k] * Bg[k, n], K=512, block tile 128(M) x 64(N), 8 waves.
__global__ __launch_bounds__(256) void gemm_qkv_kernel(const __bf16* __restrict__ W,
                                                       const __bf16* __restrict__ Xn,
                                                       const float* __restrict__ bias,
                                                       __bf16* __restrict__ Out) {
    __shared__ __bf16 Bt[2][32][64];              // double-buffered K-step B tile, 8 KB
    const int b    = blockIdx.z;
    const int wave = threadIdx.x >> 5;
    const int lane = threadIdx.x & 31;
    const int m0   = blockIdx.x * 128 + wave * 16;
    const int n0   = blockIdx.y * 64;
    const __bf16* Bb = Xn + (size_t)b * (CH * SPA);

    const int kk  = threadIdx.x >> 3;             // 0..31 : K row in tile
    const int cc8 = (threadIdx.x & 7) * 8;        // 16B chunk in N

    // prologue: stream first tile
    async_copy_b128(&Bt[0][kk][cc8], Bb + (size_t)kk * SPA + n0 + cc8);

    v8f acc[4] = {};
    int cur = 0;
    for (int k0 = 0; k0 < CH; k0 += 32) {
        if (k0 + 32 < CH) {
            // stream next tile into the other buffer while we compute on this one
            async_copy_b128(&Bt[cur ^ 1][kk][cc8],
                            Bb + (size_t)(k0 + 32 + kk) * SPA + n0 + cc8);
            wait_async1();                        // current tile done; next may be in flight
        } else {
            wait_async0();
        }
        __syncthreads();

        v16bf a = load_A(W, CH, m0, k0, lane);
        if (k0 + 32 < CH)
            __builtin_prefetch(W + (size_t)m0 * CH + k0 + 32, 0, 1);
#pragma unroll
        for (int nt = 0; nt < 4; ++nt) {
            v16bf bb;
#pragma unroll
            for (int i = 0; i < 16; ++i) bb[i] = Bt[cur][((lane & 16) >> 1) + (i & 7) + ((i & 8) << 1)][nt * 16 + (lane & 15)];
            acc[nt] = wmma_bf16(a, bb, acc[nt]);
        }
        __syncthreads();                          // all reads done before buffer refill
        cur ^= 1;
    }
    const int ncol  = lane & 15;
    const int rbase = (lane & 16) ? 8 : 0;
#pragma unroll
    for (int nt = 0; nt < 4; ++nt) {
        int s = n0 + nt * 16 + ncol;
#pragma unroll
        for (int j = 0; j < 8; ++j) {
            int o = m0 + rbase + j;
            float v = acc[nt][j] + bias[o];
            Out[((size_t)b * (3 * CH) + o) * SPA + s] = (__bf16)v;
        }
    }
}

// ---------- kernel 5: flash attention, double-buffered K/V tiles ----------
// grid (S/128, HEADS, B), 256 thr. All 8 waves share each 32-key K/V tile.
__global__ __launch_bounds__(256) void attn_kernel(const __bf16* __restrict__ QKV,
                                                   __bf16* __restrict__ Outb) {
    __shared__ __bf16 Kt[2][64][32];              // [buf][channel d][key t] 8 KB
    __shared__ __bf16 Vt[2][64][32];              // [buf][channel e][key t] 8 KB
    __shared__ __bf16 lds_p[8][16][32];           // per-wave P staging, 8 KB
    const int b    = blockIdx.z;
    const int h    = blockIdx.y;
    const int wave = threadIdx.x >> 5;
    const int lane = threadIdx.x & 31;
    const int s0   = blockIdx.x * 128 + wave * 16;

    const __bf16* Q = QKV + ((size_t)b * (3 * CH) +          h * HD) * SPA; // [HD x S]
    const __bf16* K = QKV + ((size_t)b * (3 * CH) + CH     + h * HD) * SPA;
    const __bf16* V = QKV + ((size_t)b * (3 * CH) + 2 * CH + h * HD) * SPA;

    const int rowc = threadIdx.x >> 2;            // 0..63 : channel row for staging
    const int ch8  = (threadIdx.x & 3) * 8;       // 16B chunk within 32-key row

    // prologue: stream first K/V tile
    async_copy_b128(&Kt[0][rowc][ch8], K + (size_t)rowc * SPA + ch8);
    async_copy_b128(&Vt[0][rowc][ch8], V + (size_t)rowc * SPA + ch8);

    // Q^T fragments for this wave's 16 rows, live across the whole key loop
    v16bf aq0 = load_At(Q, SPA, s0, 0,  lane);
    v16bf aq1 = load_At(Q, SPA, s0, 32, lane);

    v8f acc[4] = {};
    float rmax[8], rsum[8];
#pragma unroll
    for (int j = 0; j < 8; ++j) { rmax[j] = -1e30f; rsum[j] = 0.f; }

    const int cc = lane & 15;
    const int r0 = (lane & 16) ? 8 : 0;

    int cur = 0;
    for (int t0 = 0; t0 < SPA; t0 += 32) {
        if (t0 + 32 < SPA) {
            async_copy_b128(&Kt[cur ^ 1][rowc][ch8], K + (size_t)rowc * SPA + t0 + 32 + ch8);
            async_copy_b128(&Vt[cur ^ 1][rowc][ch8], V + (size_t)rowc * SPA + t0 + 32 + ch8);
            wait_async2();                        // current tile's 2 copies done
        } else {
            wait_async0();
        }
        __syncthreads();

        // scores: two 16x16 tiles covering keys t0..t0+31 (B frags from LDS, strided)
        v8f sc0 = {}, sc1 = {};
        v16bf bk0, bk1, bk2, bk3;
#pragma unroll
        for (int i = 0; i < 16; ++i) {
            int kd = ((lane & 16) >> 1) + (i & 7) + ((i & 8) << 1);
            bk0[i] = Kt[cur][kd][cc];
            bk1[i] = Kt[cur][32 + kd][cc];
            bk2[i] = Kt[cur][kd][16 + cc];
            bk3[i] = Kt[cur][32 + kd][16 + cc];
        }
        sc0 = wmma_bf16(aq0, bk0, sc0);
        sc0 = wmma_bf16(aq1, bk1, sc0);
        sc1 = wmma_bf16(aq0, bk2, sc1);
        sc1 = wmma_bf16(aq1, bk3, sc1);

        float p0[8], p1[8];
#pragma unroll
        for (int j = 0; j < 8; ++j) {
            float a0 = sc0[j] * 0.125f;          // 1/sqrt(hd)
            float a1 = sc1[j] * 0.125f;
            float v = fmaxf(a0, a1);
#pragma unroll
            for (int off = 8; off >= 1; off >>= 1) v = fmaxf(v, __shfl_xor(v, off, 32));
            float mnew = fmaxf(rmax[j], v);
            float corr = __expf(rmax[j] - mnew);
            rmax[j] = mnew;
            float e0 = __expf(a0 - mnew);
            float e1 = __expf(a1 - mnew);
            p0[j] = e0; p1[j] = e1;
            float rs = e0 + e1;
#pragma unroll
            for (int off = 8; off >= 1; off >>= 1) rs += __shfl_xor(rs, off, 32);
            rsum[j] = rsum[j] * corr + rs;
#pragma unroll
            for (int nt = 0; nt < 4; ++nt) acc[nt][j] *= corr;
        }

        // stage P (16x32) through wave-private LDS (D-layout -> A-layout; in-order DScnt)
#pragma unroll
        for (int j = 0; j < 8; ++j) {
            lds_p[wave][r0 + j][cc]      = (__bf16)p0[j];
            lds_p[wave][r0 + j][cc + 16] = (__bf16)p1[j];
        }
        {
            const int m  = lane & 15;
            const int kb = (lane & 16) ? 8 : 0;
            v16bf pa;
#pragma unroll
            for (int i = 0; i < 8; ++i) pa[i]     = lds_p[wave][m][kb + i];
#pragma unroll
            for (int i = 0; i < 8; ++i) pa[8 + i] = lds_p[wave][m][kb + 16 + i];

            // out += P(16x32) * V(32x64); V^T frags via strided LDS reads
#pragma unroll
            for (int nt = 0; nt < 4; ++nt) {
                v16bf bv;
#pragma unroll
                for (int i = 0; i < 16; ++i) bv[i] = Vt[cur][nt * 16 + cc][((lane & 16) >> 1) + (i & 7) + ((i & 8) << 1)];
                acc[nt] = wmma_bf16(pa, bv, acc[nt]);
            }
        }
        __syncthreads();                          // all reads done before buffer refill
        cur ^= 1;
    }

    // normalize and store channel-major bf16: out[b, h*64+e, s]
#pragma unroll
    for (int j = 0; j < 8; ++j) {
        float inv = 1.0f / rsum[j];
        int s = s0 + r0 + j;
#pragma unroll
        for (int nt = 0; nt < 4; ++nt) {
            int chn = h * HD + nt * 16 + cc;
            Outb[((size_t)b * CH + chn) * SPA + s] = (__bf16)(acc[nt][j] * inv);
        }
    }
}

// ---------- kernel 6: proj GEMM + bias + residual -> f32 ----------
__global__ __launch_bounds__(256) void gemm_proj_kernel(const __bf16* __restrict__ W,
                                                        const __bf16* __restrict__ A_in,
                                                        const float* __restrict__ bias,
                                                        const float* __restrict__ resid,
                                                        float* __restrict__ Out) {
    __shared__ __bf16 Bt[2][32][64];
    const int b    = blockIdx.z;
    const int wave = threadIdx.x >> 5;
    const int lane = threadIdx.x & 31;
    const int m0   = blockIdx.x * 128 + wave * 16;
    const int n0   = blockIdx.y * 64;
    const __bf16* Bb = A_in + (size_t)b * (CH * SPA);

    const int kk  = threadIdx.x >> 3;
    const int cc8 = (threadIdx.x & 7) * 8;

    async_copy_b128(&Bt[0][kk][cc8], Bb + (size_t)kk * SPA + n0 + cc8);

    v8f acc[4] = {};
    int cur = 0;
    for (int k0 = 0; k0 < CH; k0 += 32) {
        if (k0 + 32 < CH) {
            async_copy_b128(&Bt[cur ^ 1][kk][cc8],
                            Bb + (size_t)(k0 + 32 + kk) * SPA + n0 + cc8);
            wait_async1();
        } else {
            wait_async0();
        }
        __syncthreads();

        v16bf a = load_A(W, CH, m0, k0, lane);
        if (k0 + 32 < CH)
            __builtin_prefetch(W + (size_t)m0 * CH + k0 + 32, 0, 1);
#pragma unroll
        for (int nt = 0; nt < 4; ++nt) {
            v16bf bb;
#pragma unroll
            for (int i = 0; i < 16; ++i) bb[i] = Bt[cur][((lane & 16) >> 1) + (i & 7) + ((i & 8) << 1)][nt * 16 + (lane & 15)];
            acc[nt] = wmma_bf16(a, bb, acc[nt]);
        }
        __syncthreads();
        cur ^= 1;
    }
    const int ncol  = lane & 15;
    const int rbase = (lane & 16) ? 8 : 0;
#pragma unroll
    for (int nt = 0; nt < 4; ++nt) {
        int s = n0 + nt * 16 + ncol;
#pragma unroll
        for (int j = 0; j < 8; ++j) {
            int o = m0 + rbase + j;
            size_t idx = ((size_t)b * CH + o) * SPA + s;
            Out[idx] = acc[nt][j] + bias[o] + resid[idx];
        }
    }
}

// ---------- host launch ----------
extern "C" void kernel_launch(void* const* d_in, const int* in_sizes, int n_in,
                              void* d_out, int out_size, void* d_ws, size_t ws_size,
                              hipStream_t stream) {
    const float* x     = (const float*)d_in[0];
    const float* gnw   = (const float*)d_in[1];
    const float* gnb   = (const float*)d_in[2];
    const float* qkvw  = (const float*)d_in[3];
    const float* qkvb  = (const float*)d_in[4];
    const float* projw = (const float*)d_in[5];
    const float* projb = (const float*)d_in[6];
    float* out = (float*)d_out;

    char* ws = (char*)d_ws;
    float*  mean    = (float*)(ws + 0);                         // 256 f32
    float*  rstd    = (float*)(ws + 1024);                      // 256 f32
    __bf16* wqkv_bf = (__bf16*)(ws + 2048);                     // 1536*512
    __bf16* wprj_bf = (__bf16*)(ws + 2048 + 1572864);           // 512*512
    __bf16* xn_bf   = (__bf16*)(ws + 2099200);                  // 8*512*1024
    __bf16* qkv_bf  = (__bf16*)(ws + 10487808);                 // 8*1536*1024
    __bf16* att_bf  = (__bf16*)(ws + 35653632);                 // 8*512*1024
    (void)ws_size; (void)in_sizes; (void)n_in; (void)out_size;

    // 1) weights -> bf16
    cvt_bf16_kernel<<<(3 * CH * CH + 255) / 256, 256, 0, stream>>>(qkvw,  wqkv_bf, 3 * CH * CH);
    cvt_bf16_kernel<<<(CH * CH + 255) / 256,     256, 0, stream>>>(projw, wprj_bf, CH * CH);

    // 2) GroupNorm stats
    gn_stats_kernel<<<BATCH * GROUPS, 256, 0, stream>>>(x, mean, rstd);

    // 3) GN apply + pack bf16
    gn_apply_kernel<<<(BATCH * CH * SPA) / 256, 256, 0, stream>>>(x, mean, rstd, gnw, gnb, xn_bf);

    // 4) QKV GEMM: M=1536, N=1024, K=512 per batch
    gemm_qkv_kernel<<<dim3(12, 16, BATCH), 256, 0, stream>>>(wqkv_bf, xn_bf, qkvb, qkv_bf);

    // 5) flash attention
    attn_kernel<<<dim3(SPA / 128, HEADS, BATCH), 256, 0, stream>>>(qkv_bf, att_bf);

    // 6) proj GEMM + bias + residual (f32 output)
    gemm_proj_kernel<<<dim3(4, 16, BATCH), 256, 0, stream>>>(wprj_bf, att_bf, projb, x, out);
}
